// MyGat_21380347199683
// MI455X (gfx1250) — matrix operand order repaired and checked by reading, not compiled, not run on hardware
//
#include <hip/hip_runtime.h>
#include <stdint.h>

typedef __attribute__((ext_vector_type(16))) _Float16 v16h;
typedef __attribute__((ext_vector_type(8)))  float    v8f;
typedef __attribute__((ext_vector_type(4)))  int      v4i;

#define N_NODES 4096
#define F_IN    128
#define HEADS   4
#define F_OUT   32
#define F_TOT   128   // HEADS * F_OUT

// ---------------------------------------------------------------------------
// CDNA5 async global->LDS copy (ASYNCcnt-tracked), with builtin/asm fallback.
// Builtin signature (from clang diagnostic): (v4i AS1*, v4i AS3*, imm, imm).
// ---------------------------------------------------------------------------
typedef __attribute__((address_space(1))) v4i* gas_v4i_ptr;
typedef __attribute__((address_space(3))) v4i* lds_v4i_ptr;

__device__ __forceinline__ void async_copy_b128(void* ldsp, const void* gp) {
#if __has_builtin(__builtin_amdgcn_global_load_async_to_lds_b128)
    __builtin_amdgcn_global_load_async_to_lds_b128(
        (gas_v4i_ptr)(uintptr_t)gp,
        (lds_v4i_ptr)(uint32_t)(uintptr_t)ldsp,
        0, 0);
#else
    unsigned l = (unsigned)(uintptr_t)ldsp;   // low 32 bits = LDS byte offset
    asm volatile("global_load_async_to_lds_b128 %0, %1, off"
                 :: "v"(l), "v"(gp) : "memory");
#endif
}

#if __has_builtin(__builtin_amdgcn_s_wait_asynccnt)
#define WAIT_ASYNC(n) __builtin_amdgcn_s_wait_asynccnt(n)
#else
#define WAIT_ASYNC(n) asm volatile("s_wait_asynccnt %0" :: "i"(n) : "memory")
#endif

// ---------------------------------------------------------------------------
// Kernel 1: h' = x @ W^T  via v_wmma_f32_16x16x32_f16.
// Grid: 256 blocks (16-row tiles), 256 threads (8 waves), wave w -> cols 16w..16w+15.
// Writes h' in fp32 (for alpha kernel) and transposed f16 (h16t[col][row]) so the
// aggregation kernel's WMMA B-tiles are contiguous along K(=n).
// ---------------------------------------------------------------------------
__global__ __launch_bounds__(256)
void gat_gemm_kernel(const float* __restrict__ x, const float* __restrict__ w,
                     float* __restrict__ hp, _Float16* __restrict__ h16t) {
    const int m0   = blockIdx.x * 16;
    const int wave = threadIdx.x >> 5;
    const int lane = threadIdx.x & 31;
    const int jb   = wave * 16;            // output column tile base
    const int hi   = lane >> 4;            // lane half
    const int mrow = m0 + (lane & 15);     // A-matrix row for this lane
    const int ncol = jb + (lane & 15);     // B-matrix column for this lane

    v8f acc = {};
    for (int k0 = 0; k0 < F_IN; k0 += 32) {
        // A tile (16x32 f16): element i -> K = i + 16*hi
        v16h A;
        const float* xr = x + mrow * F_IN + k0;
        #pragma unroll
        for (int i = 0; i < 16; ++i) {
            int k = i + hi * 16;
            A[i] = (_Float16)xr[k];
        }
        // B tile (32x16 f16): B[k][n] = W[n][k0+k]; lane half selects K 0..15 / 16..31
        v16h B;
        const float* wr = w + ncol * F_IN + k0 + hi * 16;
        #pragma unroll
        for (int i = 0; i < 16; ++i) B[i] = (_Float16)wr[i];

        acc = __builtin_amdgcn_wmma_f32_16x16x32_f16(false, A, false, B,
                                                     (short)0, acc, false, false);
    }
    // C layout: VGPR v -> row (v + 8*hi), col = lane&15
    #pragma unroll
    for (int v = 0; v < 8; ++v) {
        int row = m0 + v + hi * 8;
        int col = jb + (lane & 15);
        float val = acc[v];
        hp[row * F_TOT + col]             = val;
        h16t[(size_t)col * N_NODES + row] = (_Float16)val;
    }
}

// ---------------------------------------------------------------------------
// Kernel 2: per-node attention logits alpha_src/alpha_dst (tiny).
// ---------------------------------------------------------------------------
__global__ __launch_bounds__(256)
void gat_alpha_kernel(const float* __restrict__ hp,
                      const float* __restrict__ att_src,
                      const float* __restrict__ att_dst,
                      float* __restrict__ a_src, float* __restrict__ a_dst) {
    int t = blockIdx.x * blockDim.x + threadIdx.x;
    if (t >= N_NODES * HEADS) return;
    int n = t >> 2, h = t & 3;
    const float* hr = hp + n * F_TOT + h * F_OUT;
    const float* as = att_src + h * F_OUT;
    const float* ad = att_dst + h * F_OUT;
    float s = 0.f, d = 0.f;
    #pragma unroll
    for (int f = 0; f < F_OUT; ++f) {
        float v = hr[f];
        s += v * as[f];
        d += v * ad[f];
    }
    a_src[t] = s;
    a_dst[t] = d;
}

// ---------------------------------------------------------------------------
// Kernel 3: fused masked-softmax aggregation, single pass, division deferred.
// z[m,h,f] = (sum_n adj[n,m]*exp(lrelu(as[n,h]+ad[m,h])) * h'[n,h,f]) / tau_sum[m,h] + bias
// Grid: 256 blocks (16 m-rows), 128 threads = 4 waves, wave = head.
// adj is read exactly once across the grid; tiles are double-buffered in LDS
// via ASYNCcnt-tracked global_load_async_to_lds_b128 so the copy of tile i+1
// overlaps the exp/WMMA compute of tile i.
// ---------------------------------------------------------------------------
__global__ __launch_bounds__(128)
void gat_aggregate_kernel(const float* __restrict__ adj,
                          const _Float16* __restrict__ h16t,
                          const float* __restrict__ a_src,
                          const float* __restrict__ a_dst,
                          const float* __restrict__ bias,
                          float* __restrict__ out) {
    __shared__ float sAdj[2][32][16];    // adj[n0+k][m0+m], 16B-aligned rows (async B128)
    __shared__ float sAS [2][32][HEADS]; // alpha_src[n0+k][h]

    const int m0   = blockIdx.x * 16;
    const int h    = threadIdx.x >> 5;     // head = wave id
    const int lane = threadIdx.x & 31;
    const int mloc = lane & 15;            // row m for A / col f for C
    const int hi   = lane >> 4;

    const float ad = a_dst[(m0 + mloc) * HEADS + h];

    // B-tile base pointers (h16t is [F_TOT][N_NODES], K contiguous):
    const _Float16* b0 = h16t + (size_t)(h * F_OUT + mloc) * N_NODES + hi * 16;
    const _Float16* b1 = b0 + (size_t)16 * N_NODES;

    const int tk = threadIdx.x >> 2;  // 0..31 : adj row within tile
    const int tg = threadIdx.x & 3;   // group of 4 columns / head index

    v8f acc0 = {}, acc1 = {};
    float ts = 0.f;

    // Prologue: stage tile 0 into buffer 0.
    async_copy_b128(&sAdj[0][tk][tg * 4],
                    adj + (size_t)tk * N_NODES + m0 + tg * 4);
    sAS[0][tk][tg] = a_src[tk * HEADS + tg];

    const int NIT = N_NODES / 32;
    for (int it = 0; it < NIT; ++it) {
        const int n0  = it * 32;
        const int buf = it & 1;

        __syncthreads();                  // all waves done reading buf^1 (iter it-1)
        if (it + 1 < NIT) {               // stage tile it+1 into buf^1
            const int nn = n0 + 32;
            async_copy_b128(&sAdj[buf ^ 1][tk][tg * 4],
                            adj + (size_t)(nn + tk) * N_NODES + m0 + tg * 4);
            sAS[buf ^ 1][tk][tg] = a_src[(nn + tk) * HEADS + tg];
            WAIT_ASYNC(1);                // tile it landed (in-order completion)
        } else {
            WAIT_ASYNC(0);                // last tile landed
        }
        __syncthreads();                  // tile it landed for every wave

        // Build f16 A-tile in WMMA lane layout; accumulate tau partial on the fly.
        v16h A;
        #pragma unroll
        for (int i = 0; i < 16; ++i) {
            int k   = i + hi * 16;
            float c = sAS[buf][k][h] + ad;
            float e = __expf(fmaxf(0.2f * c, c));   // exp(leaky_relu)
            float a = sAdj[buf][k][mloc] * e;       // adjacency mask (binary)
            A[i] = (_Float16)a;
            ts  += a;
        }
        v16h B0 = *(const v16h*)(b0 + n0);          // 32B aligned contiguous K
        v16h B1 = *(const v16h*)(b1 + n0);
        acc0 = __builtin_amdgcn_wmma_f32_16x16x32_f16(false, A, false, B0,
                                                      (short)0, acc0, false, false);
        acc1 = __builtin_amdgcn_wmma_f32_16x16x32_f16(false, A, false, B1,
                                                      (short)0, acc1, false, false);
    }

    // tau_sum[m = lane&15]: lane and lane+16 hold disjoint K halves.
    ts += __shfl_xor(ts, 16, 32);

    const float* bb = bias + h * F_OUT;
    const float  bA = bb[mloc];
    const float  bB = bb[16 + mloc];
    #pragma unroll
    for (int v = 0; v < 8; ++v) {
        int   row  = v + hi * 8;                 // C row for this VGPR
        float tsum = __shfl(ts, row, 32);        // tau_sum of that m-row
        float inv  = 1.0f / tsum;
        int   grow = m0 + row;
        out[grow * F_TOT + h * F_OUT + mloc]      = acc0[v] * inv + bA;
        out[grow * F_TOT + h * F_OUT + 16 + mloc] = acc1[v] * inv + bB;
    }
}

// ---------------------------------------------------------------------------
// Host-side launcher
// ---------------------------------------------------------------------------
extern "C" void kernel_launch(void* const* d_in, const int* in_sizes, int n_in,
                              void* d_out, int out_size, void* d_ws, size_t ws_size,
                              hipStream_t stream) {
    const float* x       = (const float*)d_in[0];   // [4096,128]
    const float* adj     = (const float*)d_in[1];   // [4096,4096]
    const float* weight  = (const float*)d_in[2];   // [128,128]
    const float* att_src = (const float*)d_in[3];   // [1,4,32]
    const float* att_dst = (const float*)d_in[4];   // [1,4,32]
    const float* bias    = (const float*)d_in[5];   // [128]
    float*       out     = (float*)d_out;           // [4096,128]

    // Workspace partition: h' fp32 (2MB) | h16t (1MB) | alpha_src (64KB) | alpha_dst (64KB)
    char* ws = (char*)d_ws;
    float*    hp    = (float*)ws;    ws += (size_t)N_NODES * F_TOT * sizeof(float);
    _Float16* h16t  = (_Float16*)ws; ws += (size_t)N_NODES * F_TOT * sizeof(_Float16);
    float*    a_src = (float*)ws;    ws += (size_t)N_NODES * HEADS * sizeof(float);
    float*    a_dst = (float*)ws;

    gat_gemm_kernel<<<N_NODES / 16, 256, 0, stream>>>(x, weight, hp, h16t);
    gat_alpha_kernel<<<(N_NODES * HEADS + 255) / 256, 256, 0, stream>>>(hp, att_src, att_dst, a_src, a_dst);
    gat_aggregate_kernel<<<N_NODES / 16, 128, 0, stream>>>(adj, h16t, a_src, a_dst, bias, out);
}
